// CausalSelfAttention_69827578298797
// MI455X (gfx1250) — matrix operand order repaired and checked by reading, not compiled, not run on hardware
//
#include <hip/hip_runtime.h>
#include <hip/hip_bf16.h>

// CDNA5 / gfx1250 causal self-attention.
// Pipeline: f32->f16 converts -> TDM-fed double-buffered WMMA GEMMs (QKV) ->
//           flash attention (WMMA) -> TDM-fed WMMA output projection.

typedef __attribute__((ext_vector_type(16))) _Float16 v16h;
typedef __attribute__((ext_vector_type(8)))  _Float16 v8h;
typedef __attribute__((ext_vector_type(8)))  float    v8f;
typedef __attribute__((ext_vector_type(4)))  unsigned int u32x4;
typedef __attribute__((ext_vector_type(8)))  int      i32x8;
typedef __attribute__((ext_vector_type(4)))  int      i32x4;

union HV { v16h v; v8h h[2]; };
__device__ __forceinline__ v16h cat8(v8h lo, v8h hi) { HV u; u.h[0] = lo; u.h[1] = hi; return u.v; }
__device__ __forceinline__ v8f zerov8() { v8f z = {0.f,0.f,0.f,0.f,0.f,0.f,0.f,0.f}; return z; }

#define WMMA32(A,B,C) __builtin_amdgcn_wmma_f32_16x16x32_f16(false,(A),false,(B),(short)0,(C),false,false)

constexpr int kB = 4, kT = 2048, kC = 1024, kH = 16, kD = 64;
constexpr int kM = kB * kT;  // 8192 tokens

// ---------------------------------------------------------------------------
// f32 -> f16 elementwise convert (8 elems / thread)
// ---------------------------------------------------------------------------
__global__ __launch_bounds__(256) void f32_to_f16_kernel(
    const float* __restrict__ in, _Float16* __restrict__ out, int n)
{
  const int i = (blockIdx.x * blockDim.x + threadIdx.x) * 8;
  if (i + 8 <= n) {
    const float4 a = *(const float4*)(in + i);
    const float4 b = *(const float4*)(in + i + 4);
    v8h h = {(_Float16)a.x,(_Float16)a.y,(_Float16)a.z,(_Float16)a.w,
             (_Float16)b.x,(_Float16)b.y,(_Float16)b.z,(_Float16)b.w};
    *(v8h*)(out + i) = h;
  }
}

// ---------------------------------------------------------------------------
// TDM: DMA a (nrows x 32) f16 tile (row stride kC elems in memory) into LDS.
// D# per CDNA5 ISA:
//   data_size=2B; pad every 16 DWORDs (one 32-half row) by 4 DWORDs (8 halves)
//   -> LDS row stride = 40 halves, matching the WMMA fragment loaders.
// ---------------------------------------------------------------------------
__device__ __forceinline__ void tdm_load_tile(void* lds, const _Float16* gsrc,
                                              int row0, int nrows, int col0,
                                              int tot_rows)
{
  const unsigned long long ga =
      (unsigned long long)(uintptr_t)(gsrc + (size_t)row0 * kC + col0);
  u32x4 g0;
  g0[0] = 1u;                                             // count=1, user D#
  g0[1] = (unsigned)(uintptr_t)lds;                       // lds_addr (bytes)
  g0[2] = (unsigned)(ga & 0xffffffffu);                   // global_addr[31:0]
  g0[3] = (unsigned)((ga >> 32) & 0x1ffffffu) | (2u << 30); // addr[56:32]|type=2
  i32x8 g1;
  g1[0] = (int)((1u << 16) | (1u << 20) | (3u << 22) | (3u << 25));
  //            data_size=2B  pad_enable  interval=16DW  amount=4DW
  g1[1] = (int)(((unsigned)kC & 0xffffu) << 16);          // tensor_dim0 lo16
  g1[2] = (int)((((unsigned)kC) >> 16) | (((unsigned)tot_rows & 0xffffu) << 16));
  g1[3] = (int)((((unsigned)tot_rows) >> 16) | (32u << 16)); // tile_dim0=32
  g1[4] = (int)(unsigned)nrows;                           // tile_dim1 (dim2=0)
  g1[5] = (int)(unsigned)kC;                              // tensor_dim0_stride
  g1[6] = 0;
  g1[7] = 0;
  i32x4 z4 = {0, 0, 0, 0};
#if defined(__clang_major__) && __clang_major__ >= 23
  i32x8 z8 = {0, 0, 0, 0, 0, 0, 0, 0};
  __builtin_amdgcn_tensor_load_to_lds(g0, g1, z4, z4, z8, 0);
#else
  __builtin_amdgcn_tensor_load_to_lds(g0, g1, z4, z4, 0);
#endif
}

// ---------------------------------------------------------------------------
// GEMM: Out[m,n] = A[m,:] . W[n,:] + bias[n]  (A: 8192x1024 f16, W: 1024x1024 f16)
// Block tile 128x64, 8 waves, wave = 32x32 micro-tile (2x2 WMMA).
// A/W tiles staged by TDM, double-buffered: issue k-step s+1, wait cnt<=2,
// compute k-step s from the other buffer.
// MODE 0: f16 out (B,H,T,D); MODE 1: f16 out (B,H,D,T); MODE 2: f32 out (M,C).
// ---------------------------------------------------------------------------
template<int MODE>
__global__ __launch_bounds__(256) void gemm_tdm_kernel(
    const _Float16* __restrict__ A, const _Float16* __restrict__ Wt,
    const float* __restrict__ bias, void* __restrict__ Outp)
{
  __shared__ alignas(32) _Float16 lA[2][128 * 40];
  __shared__ alignas(32) _Float16 lW[2][64 * 40];

  const int tid   = threadIdx.x;
  const int lane  = tid & 31, wid = tid >> 5;
  const int l15   = lane & 15, hhalf = lane >> 4;
  const int wm    = wid & 3, wn = wid >> 2;      // waves: 4 (M) x 2 (N)
  const int bM    = blockIdx.x, bN = blockIdx.y;
  const int kb8   = hhalf << 3;
  const int kb16  = hhalf << 4;

  v8f acc[2][2];
  #pragma unroll
  for (int i = 0; i < 2; ++i)
    #pragma unroll
    for (int j = 0; j < 2; ++j) acc[i][j] = zerov8();

  constexpr int NSTEP = kC / 32;  // 32
  if (tid < 32) {  // wave 0 drives the TDM (EXEC-independent, TENSORcnt-tracked)
    tdm_load_tile(&lA[0][0], A,  bM * 128, 128, 0, kM);
    tdm_load_tile(&lW[0][0], Wt, bN * 64,  64,  0, kC);
  }

  for (int s = 0; s < NSTEP; ++s) {
    const int buf = s & 1;
    if (tid < 32) {
      if (s + 1 < NSTEP) {
        tdm_load_tile(&lA[buf ^ 1][0], A,  bM * 128, 128, (s + 1) * 32, kM);
        tdm_load_tile(&lW[buf ^ 1][0], Wt, bN * 64,  64,  (s + 1) * 32, kC);
        __builtin_amdgcn_s_wait_tensorcnt(2);  // buffer `buf` landed
      } else {
        __builtin_amdgcn_s_wait_tensorcnt(0);
      }
    }
    __syncthreads();

    v16h af[2], bf[2];
    #pragma unroll
    for (int i = 0; i < 2; ++i) {
      const _Float16* p = &lA[buf][(wm * 32 + i * 16 + l15) * 40];
      af[i] = cat8(*(const v8h*)(p + kb16), *(const v8h*)(p + kb16 + 8));
    }
    #pragma unroll
    for (int j = 0; j < 2; ++j) {
      const _Float16* p = &lW[buf][(wn * 32 + j * 16 + l15) * 40] + kb16;
      bf[j] = cat8(*(const v8h*)(p), *(const v8h*)(p + 8));
    }
    #pragma unroll
    for (int i = 0; i < 2; ++i)
      #pragma unroll
      for (int j = 0; j < 2; ++j)
        acc[i][j] = WMMA32(af[i], bf[j], acc[i][j]);
    __syncthreads();  // reads done before TDM overwrites this buffer
  }

  // ---- epilogue: bias + layout-specific store ----
  #pragma unroll
  for (int i = 0; i < 2; ++i) {
    #pragma unroll
    for (int j = 0; j < 2; ++j) {
      const int gn = bN * 64 + wn * 32 + j * 16 + l15;
      const float bvv = bias[gn];
      #pragma unroll
      for (int v = 0; v < 8; ++v) {
        const int gm = bM * 128 + wm * 32 + i * 16 + v + (hhalf << 3);
        const float val = acc[i][j][v] + bvv;
        if (MODE == 2) {
          ((float*)Outp)[(size_t)gm * kC + gn] = val;
        } else {
          const int b = gm >> 11, t = gm & (kT - 1);
          const int h = gn >> 6, d = gn & 63;
          if (MODE == 0)
            ((_Float16*)Outp)[(((size_t)b * kH + h) * kT + t) * kD + d] = (_Float16)val;
          else
            ((_Float16*)Outp)[(((size_t)b * kH + h) * kD + d) * kT + t] = (_Float16)val;
        }
      }
    }
  }
}

// ---------------------------------------------------------------------------
// Flash attention: block = (b,h, 128 q-rows); wave owns 16 q-rows.
// Per 32-key chunk: 4 WMMAs for S = Q K^T, online softmax, P C->A layout via
// per-wave LDS, 4 WMMAs for O += P V (V stored transposed).
// ---------------------------------------------------------------------------
__global__ __launch_bounds__(256) void attn_kernel(
    const _Float16* __restrict__ Q, const _Float16* __restrict__ K,
    const _Float16* __restrict__ Vt, _Float16* __restrict__ Y)
{
  __shared__ alignas(32) _Float16 lP[8][16 * 32];

  const int tid = threadIdx.x, lane = tid & 31, wid = tid >> 5;
  const int l15 = lane & 15, hhalf = lane >> 4;
  const int kb8 = hhalf << 3, kb16 = hhalf << 4;
  const int bh = blockIdx.y;
  const int qrow0 = blockIdx.x * 128 + wid * 16;
  const _Float16* Qh = Q  + (size_t)bh * kT * kD;
  const _Float16* Kh = K  + (size_t)bh * kT * kD;
  const _Float16* Vh = Vt + (size_t)bh * kD * kT;
  _Float16* myP = &lP[wid][0];

  v16h aQ[2];
  #pragma unroll
  for (int s2 = 0; s2 < 2; ++s2) {
    const _Float16* p = Qh + (size_t)(qrow0 + l15) * kD + 32 * s2;
    aQ[s2] = cat8(*(const v8h*)(p + kb16), *(const v8h*)(p + kb16 + 8));
  }

  v8f o[4];
  #pragma unroll
  for (int dt = 0; dt < 4; ++dt) o[dt] = zerov8();
  float mrow[8], lrow[8];
  #pragma unroll
  for (int v = 0; v < 8; ++v) { mrow[v] = -1e30f; lrow[v] = 0.f; }

  const float rscale = 0.125f;                 // 1/sqrt(64)
  const int nch = (qrow0 + 47) >> 5;           // ceil((qrow0+16)/32)

  for (int ch = 0; ch < nch; ++ch) {
    const int cb = ch << 5;
    if (ch + 1 < nch) {                        // global_prefetch_b8
      __builtin_prefetch(Kh + (size_t)(cb + 32 + l15) * kD, 0, 1);
      __builtin_prefetch(Vh + (size_t)l15 * kT + cb + 32, 0, 1);
    }

    v8f s[2];
    #pragma unroll
    for (int j = 0; j < 2; ++j) {
      const _Float16* krow = Kh + (size_t)(cb + j * 16 + l15) * kD;
      const v16h b0 = *(const v16h*)(krow + kb16);
      const v16h b1 = *(const v16h*)(krow + 32 + kb16);
      s[j] = WMMA32(aQ[0], b0, zerov8());
      s[j] = WMMA32(aQ[1], b1, s[j]);
    }

    #pragma unroll
    for (int j = 0; j < 2; ++j)
      #pragma unroll
      for (int v = 0; v < 8; ++v) {
        const int sg = cb + j * 16 + l15;
        const int qg = qrow0 + v + (hhalf << 3);
        const float val = s[j][v] * rscale;
        s[j][v] = (sg <= qg) ? val : -1e30f;
      }

    #pragma unroll
    for (int v = 0; v < 8; ++v) {
      float t = fmaxf(s[0][v], s[1][v]);
      #pragma unroll
      for (int off = 8; off > 0; off >>= 1) t = fmaxf(t, __shfl_xor(t, off, 32));
      const float mn = fmaxf(mrow[v], t);
      const float alpha = __expf(mrow[v] - mn);
      mrow[v] = mn;
      const float p0 = __expf(s[0][v] - mn);
      const float p1 = __expf(s[1][v] - mn);
      s[0][v] = p0; s[1][v] = p1;
      float rs = p0 + p1;
      #pragma unroll
      for (int off = 8; off > 0; off >>= 1) rs += __shfl_xor(rs, off, 32);
      lrow[v] = lrow[v] * alpha + rs;
      #pragma unroll
      for (int dt = 0; dt < 4; ++dt) o[dt][v] *= alpha;
    }

    #pragma unroll
    for (int j = 0; j < 2; ++j)
      #pragma unroll
      for (int v = 0; v < 8; ++v)
        myP[(v + (hhalf << 3)) * 32 + j * 16 + l15] = (_Float16)s[j][v];
    const _Float16* pp = myP + l15 * 32;
    const v16h aP = cat8(*(const v8h*)(pp + kb16), *(const v8h*)(pp + kb16 + 8));

    #pragma unroll
    for (int dt = 0; dt < 4; ++dt) {
      const v16h bV = *(const v16h*)(Vh + (size_t)(dt * 16 + l15) * kT + cb + kb16);
      o[dt] = WMMA32(aP, bV, o[dt]);
    }
  }

  const int b = bh >> 4, h = bh & 15;
  float inv[8];
  #pragma unroll
  for (int v = 0; v < 8; ++v) inv[v] = 1.0f / lrow[v];
  #pragma unroll
  for (int dt = 0; dt < 4; ++dt)
    #pragma unroll
    for (int v = 0; v < 8; ++v) {
      const int t = qrow0 + v + (hhalf << 3);
      Y[((size_t)b * kT + t) * kC + h * 64 + dt * 16 + l15] = (_Float16)(o[dt][v] * inv[v]);
    }
}

// ---------------------------------------------------------------------------
extern "C" void kernel_launch(void* const* d_in, const int* in_sizes, int n_in,
                              void* d_out, int out_size, void* d_ws, size_t ws_size,
                              hipStream_t stream)
{
  const float* x  = (const float*)d_in[0];
  const float* Wk = (const float*)d_in[1];
  const float* bk = (const float*)d_in[2];
  const float* Wq = (const float*)d_in[3];
  const float* bq = (const float*)d_in[4];
  const float* Wv = (const float*)d_in[5];
  const float* bv = (const float*)d_in[6];
  const float* Wp = (const float*)d_in[7];
  const float* bp = (const float*)d_in[8];

  const size_t eTok = (size_t)kM * kC;   // 8.39M elems
  const size_t eW   = (size_t)kC * kC;   // 1.05M elems
  _Float16* x16  = (_Float16*)d_ws;      // 16 MiB (reused as Ym later)
  _Float16* w16q = x16  + eTok;
  _Float16* w16k = w16q + eW;
  _Float16* w16v = w16k + eW;
  _Float16* w16p = w16v + eW;
  _Float16* Qf   = w16p + eW;            // (B,H,T,D)
  _Float16* Kf   = Qf   + eTok;          // (B,H,T,D)
  _Float16* Vtf  = Kf   + eTok;          // (B,H,D,T)
  _Float16* Ym   = x16;                  // alias: x16 dead after QKV GEMMs
  // requires ws_size >= 72 MiB

  const dim3 blk(256);
  f32_to_f16_kernel<<<dim3((unsigned)(eTok / 2048)), blk, 0, stream>>>(x,  x16,  (int)eTok);
  f32_to_f16_kernel<<<dim3((unsigned)(eW   / 2048)), blk, 0, stream>>>(Wq, w16q, (int)eW);
  f32_to_f16_kernel<<<dim3((unsigned)(eW   / 2048)), blk, 0, stream>>>(Wk, w16k, (int)eW);
  f32_to_f16_kernel<<<dim3((unsigned)(eW   / 2048)), blk, 0, stream>>>(Wv, w16v, (int)eW);
  f32_to_f16_kernel<<<dim3((unsigned)(eW   / 2048)), blk, 0, stream>>>(Wp, w16p, (int)eW);

  const dim3 gg(kM / 128, kC / 64);      // (64, 16)
  gemm_tdm_kernel<0><<<gg, blk, 0, stream>>>(x16, w16q, bq, Qf);
  gemm_tdm_kernel<0><<<gg, blk, 0, stream>>>(x16, w16k, bk, Kf);
  gemm_tdm_kernel<1><<<gg, blk, 0, stream>>>(x16, w16v, bv, Vtf);
  attn_kernel<<<dim3(kT / 128, kB * kH), blk, 0, stream>>>(Qf, Kf, Vtf, Ym);
  gemm_tdm_kernel<2><<<gg, blk, 0, stream>>>(Ym, w16p, bp, (float*)d_out);
}
